// MultiheadAttentionForLSHSort_41489384079580
// MI455X (gfx1250) — matrix operand order repaired and checked by reading, not compiled, not run on hardware
//
#include <hip/hip_runtime.h>
#include <hip/hip_bf16.h>
#include <math.h>

typedef __attribute__((ext_vector_type(16))) __bf16       bf16x16;
typedef __attribute__((ext_vector_type(8)))  float        f32x8;
typedef __attribute__((ext_vector_type(4)))  unsigned int u32x4;

#define BM 128
#define BN 128
#define BK 32
#define LDSS 40        // BK + 8 pad, keeps rows 16B aligned (40*2 = 80B)
#define HDP 136        // head-tile LDS stride (128 + 8), 16B aligned
#define NTHREADS 256   // 8 wave32 waves

union Frag { bf16x16 v; u32x4 q[2]; __bf16 h[16]; };

// ---------------------------------------------------------------------------
// Stage one 128x32 A tile and one 128x32 W tile into LDS (bf16).
// 256 threads: thread t -> row t>>1, 16-element half (t&1).
// ---------------------------------------------------------------------------
template<bool A_F32>
__device__ __forceinline__ void load_stage(
    const void* __restrict__ Ap, const __bf16* __restrict__ W,
    long lda, long ldw, int bmBase, int bnBase, int kcol,
    __bf16* __restrict__ As, __bf16* __restrict__ Bs)
{
  const int t   = threadIdx.x;
  const int row = t >> 1;
  const int col = (t & 1) * 16;

  const __bf16* w = W + (long)(bnBase + row) * ldw + kcol + col;
  __builtin_prefetch((const void*)(w + 2 * BK), 0, 1);   // global_prefetch_b8
  u32x4 w0 = *(const u32x4*)w;
  u32x4 w1 = *(const u32x4*)(w + 8);

  if (A_F32) {
    const float* a = (const float*)Ap + (long)(bmBase + row) * lda + kcol + col;
    Frag f;
#pragma unroll
    for (int i = 0; i < 16; ++i) f.h[i] = (__bf16)a[i];
    *(u32x4*)&As[row * LDSS + col]     = f.q[0];
    *(u32x4*)&As[row * LDSS + col + 8] = f.q[1];
  } else {
    const __bf16* a = (const __bf16*)Ap + (long)(bmBase + row) * lda + kcol + col;
    *(u32x4*)&As[row * LDSS + col]     = *(const u32x4*)a;
    *(u32x4*)&As[row * LDSS + col + 8] = *(const u32x4*)(a + 8);
  }
  *(u32x4*)&Bs[row * LDSS + col]     = w0;
  *(u32x4*)&Bs[row * LDSS + col + 8] = w1;
}

// ---------------------------------------------------------------------------
// C[128x128] += A[128xK] * W[128xK]^T  using v_wmma_f32_16x16x32_bf16.
// 8 waves in a 2(M) x 4(N) grid; each wave owns 64x32 (4x2 16x16 tiles).
// LDS double-buffered; one __syncthreads per K-step.
// Fragment layouts as loaded here (same K order in A and B):
//  A 16x32: lane&15 = M, lane>>4 selects K 0..15 vs 16..31, elem e = +K.
//  B 32x16: lane&15 = N, lane>>4 selects K 0..15 vs 16..31, elem e = +K.
//  C 16x16 f32: elem r -> M = r + (lane>>4)*8, N = lane&15.
// ---------------------------------------------------------------------------
template<bool A_F32>
__device__ __forceinline__ void gemm_accum(
    const void* __restrict__ A, const __bf16* __restrict__ W, int K,
    int bmBase, int bnBase, __bf16* __restrict__ As, __bf16* __restrict__ Bs,
    f32x8 acc[4][2])
{
  const int lane = threadIdx.x & 31;
  const int wave = threadIdx.x >> 5;
  const int wm = (wave & 1) * 64;
  const int wn = (wave >> 1) * 32;
  const int ln = lane & 15;
  const int kh = lane >> 4;
  const int nk = K / BK;

  load_stage<A_F32>(A, W, K, K, bmBase, bnBase, 0, As, Bs);
  __syncthreads();

#pragma unroll 1
  for (int kb = 0; kb < nk; ++kb) {
    const int cur = kb & 1;
    const __bf16* Ac = As + cur * (BM * LDSS);
    const __bf16* Bc = Bs + cur * (BM * LDSS);
    if (kb + 1 < nk)
      load_stage<A_F32>(A, W, K, K, bmBase, bnBase, (kb + 1) * BK,
                        As + (cur ^ 1) * (BM * LDSS),
                        Bs + (cur ^ 1) * (BM * LDSS));

    Frag fb[2];
#pragma unroll
    for (int j = 0; j < 2; ++j) {
      const __bf16* bp = &Bc[(wn + j * 16 + ln) * LDSS + kh * 16];
      fb[j].q[0] = *(const u32x4*)bp;
      fb[j].q[1] = *(const u32x4*)(bp + 8);
    }
#pragma unroll
    for (int i = 0; i < 4; ++i) {
      Frag fa;
      const __bf16* ap = &Ac[(wm + i * 16 + ln) * LDSS + kh * 16];
      fa.q[0] = *(const u32x4*)ap;
      fa.q[1] = *(const u32x4*)(ap + 8);
#pragma unroll
      for (int j = 0; j < 2; ++j) {
        acc[i][j] = __builtin_amdgcn_wmma_f32_16x16x32_bf16(
            false, fa.v, false, fb[j].v, (short)0, acc[i][j], false, false);
      }
    }
    __syncthreads();
  }
}

// ---------------------------------------------------------------------------
// C = A @ W^T + bias.  A: [M,K] (f32 or bf16), W: [N,K] bf16 row-major.
// ---------------------------------------------------------------------------
template<bool A_F32, bool OUT_BF16>
__global__ __launch_bounds__(NTHREADS)
void gemm_xwt(const void* __restrict__ A, const __bf16* __restrict__ W,
              const float* __restrict__ bias, void* __restrict__ C,
              int N, int K)
{
  extern __shared__ char smem[];
  __bf16* As = (__bf16*)smem;
  __bf16* Bs = As + 2 * BM * LDSS;

  f32x8 acc[4][2];
#pragma unroll
  for (int i = 0; i < 4; ++i)
#pragma unroll
    for (int j = 0; j < 2; ++j)
#pragma unroll
      for (int e = 0; e < 8; ++e) acc[i][j][e] = 0.f;

  const int bm = blockIdx.x * BM;
  const int bn = blockIdx.y * BN;
  gemm_accum<A_F32>(A, W, K, bm, bn, As, Bs, acc);

  const int lane = threadIdx.x & 31;
  const int wave = threadIdx.x >> 5;
  const int wm = (wave & 1) * 64;
  const int wn = (wave >> 1) * 32;
  const int ln = lane & 15;
  const int kh = lane >> 4;

#pragma unroll
  for (int i = 0; i < 4; ++i) {
#pragma unroll
    for (int j = 0; j < 2; ++j) {
      const int col = bn + wn + j * 16 + ln;
      const float bb = bias ? bias[col] : 0.f;
#pragma unroll
      for (int r = 0; r < 8; ++r) {
        const long row = (long)bm + wm + i * 16 + kh * 8 + r;
        const float v = acc[i][j][r] + bb;
        if (OUT_BF16) ((__bf16*)C)[row * (long)N + col] = (__bf16)v;
        else          ((float*)C)[row * (long)N + col] = v;
      }
    }
  }
}

// ---------------------------------------------------------------------------
// Fused per-head q/k/v projection + segment(4)-local masked softmax attention.
// Block = 128 rows x 1 head. q/k/v head tiles (128x128 bf16) live in LDS.
// Wq/bq are pre-scaled by 1/sqrt(hd). o layout: [M, 1024], head h at col h*128.
// ---------------------------------------------------------------------------
__global__ __launch_bounds__(NTHREADS)
void qkv_attn(const __bf16* __restrict__ qkin, const __bf16* __restrict__ vin,
              const __bf16* __restrict__ wq, const __bf16* __restrict__ wk,
              const __bf16* __restrict__ wv2,
              const float* __restrict__ bq, const float* __restrict__ bk,
              const float* __restrict__ bv,
              __bf16* __restrict__ o, int K)
{
  extern __shared__ char smem[];
  __bf16* As    = (__bf16*)smem;
  __bf16* Bs    = As + 2 * BM * LDSS;
  __bf16* tiles = Bs + 2 * BM * LDSS;   // 3 x [128][HDP] bf16

  const int bm = blockIdx.x * BM;
  const int h  = blockIdx.y;
  const int lane = threadIdx.x & 31;
  const int wave = threadIdx.x >> 5;
  const int wm = (wave & 1) * 64;
  const int wn = (wave >> 1) * 32;
  const int ln = lane & 15;
  const int kh = lane >> 4;

#pragma unroll 1
  for (int p = 0; p < 3; ++p) {
    const __bf16* Ain = (p == 2) ? vin : qkin;
    const __bf16* Wp  = (p == 0) ? wq : (p == 1) ? wk : wv2;
    const float*  bp  = (p == 0) ? bq : (p == 1) ? bk : bv;
    __bf16* T = tiles + p * (BM * HDP);

    f32x8 acc[4][2];
#pragma unroll
    for (int i = 0; i < 4; ++i)
#pragma unroll
      for (int j = 0; j < 2; ++j)
#pragma unroll
        for (int e = 0; e < 8; ++e) acc[i][j][e] = 0.f;

    gemm_accum<false>((const void*)Ain, Wp, K, bm, h * 128, As, Bs, acc);

#pragma unroll
    for (int i = 0; i < 4; ++i)
#pragma unroll
      for (int j = 0; j < 2; ++j) {
        const int cl = wn + j * 16 + ln;
        const float bb = bp[h * 128 + cl];
#pragma unroll
        for (int r = 0; r < 8; ++r) {
          const int rl = wm + i * 16 + kh * 8 + r;
          T[rl * HDP + cl] = (__bf16)(acc[i][j][r] + bb);
        }
      }
  }
  __syncthreads();

  if (threadIdx.x < BM) {
    const int r  = threadIdx.x;
    const int s  = r & ~3;     // segment base row
    const int ii = r & 3;      // position in segment (diagonal to mask)
    const __bf16* qt = tiles;
    const __bf16* kt = tiles + BM * HDP;
    const __bf16* vt = tiles + 2 * BM * HDP;
    const __bf16* qr = qt + r * HDP;

    float sc[4];
#pragma unroll
    for (int j = 0; j < 4; ++j) {
      const __bf16* kr = kt + (s + j) * HDP;
      float d = 0.f;
      for (int e = 0; e < 128; ++e) d += (float)qr[e] * (float)kr[e];
      sc[j] = (j == ii) ? -3.0e38f : d;
    }
    const float mx = fmaxf(fmaxf(sc[0], sc[1]), fmaxf(sc[2], sc[3]));
    float ex[4], sum = 0.f;
#pragma unroll
    for (int j = 0; j < 4; ++j) { ex[j] = __expf(sc[j] - mx); sum += ex[j]; }
    const float inv = 1.f / sum;

    __bf16* orow = o + (long)(bm + r) * 1024 + h * 128;
    for (int e = 0; e < 128; ++e) {
      float a = 0.f;
#pragma unroll
      for (int j = 0; j < 4; ++j) a += ex[j] * (float)vt[(s + j) * HDP + e];
      orow[e] = (__bf16)(a * inv);
    }
  }
}

// ---------------------------------------------------------------------------
// Small elementwise helpers.
// ---------------------------------------------------------------------------
__global__ void cast_f32_to_bf16(const float* __restrict__ in,
                                 __bf16* __restrict__ out, long n, float scale)
{
  long i = (long)blockIdx.x * blockDim.x + threadIdx.x;
  if (i < n) out[i] = (__bf16)(in[i] * scale);
}

__global__ void scale_f32(const float* __restrict__ in, float* __restrict__ out,
                          long n, float scale)
{
  long i = (long)blockIdx.x * blockDim.x + threadIdx.x;
  if (i < n) out[i] = in[i] * scale;
}

// ---------------------------------------------------------------------------
extern "C" void kernel_launch(void* const* d_in, const int* in_sizes, int n_in,
                              void* d_out, int out_size, void* d_ws, size_t ws_size,
                              hipStream_t stream)
{
  const int  D = 1024;
  const int  H = 8;
  const float qscale = 0.0883883476483184f;   // 1 / sqrt(128)

  const float* x   = (const float*)d_in[0];
  const float* wqk = (const float*)d_in[1];
  const float* bqk = (const float*)d_in[2];
  const float* wv  = (const float*)d_in[3];
  const float* bvp = (const float*)d_in[4];
  const float* win = (const float*)d_in[5];   // [3D, D]
  const float* bin = (const float*)d_in[6];   // [3D]
  const float* wo  = (const float*)d_in[7];
  const float* bo  = (const float*)d_in[8];

  const long M = (long)in_sizes[0] / D;       // 32768 rows (permutation cancels)

  // ---- workspace layout ----
  char* ws = (char*)d_ws;
  const size_t WB   = (size_t)D * D * sizeof(__bf16);       // 2 MB per weight
  __bf16* wqk16 = (__bf16*)(ws + 0 * WB);
  __bf16* wv16  = (__bf16*)(ws + 1 * WB);
  __bf16* wq16  = (__bf16*)(ws + 2 * WB);     // pre-scaled by qscale
  __bf16* wk16  = (__bf16*)(ws + 3 * WB);
  __bf16* wv216 = (__bf16*)(ws + 4 * WB);
  __bf16* wo16  = (__bf16*)(ws + 5 * WB);
  float*  bq_s  = (float*)(ws + 6 * WB);      // pre-scaled bias (4 KB)
  const size_t base = 6 * WB + 65536;
  const size_t MDB  = (size_t)M * D * sizeof(__bf16);       // 64 MB
  __bf16* qkin  = (__bf16*)(ws + base);
  __bf16* vin16 = (__bf16*)(ws + base + MDB);
  __bf16* o16   = (__bf16*)(ws + base + 2 * MDB);

  // ---- weight casts (bf16, scale folded into Wq/bq) ----
  const long nW = (long)D * D;
  dim3 cb(256), cgW((unsigned)((nW + 255) / 256));
  cast_f32_to_bf16<<<cgW, cb, 0, stream>>>(wqk,          wqk16, nW, 1.f);
  cast_f32_to_bf16<<<cgW, cb, 0, stream>>>(wv,           wv16,  nW, 1.f);
  cast_f32_to_bf16<<<cgW, cb, 0, stream>>>(win,          wq16,  nW, qscale);
  cast_f32_to_bf16<<<cgW, cb, 0, stream>>>(win + nW,     wk16,  nW, 1.f);
  cast_f32_to_bf16<<<cgW, cb, 0, stream>>>(win + 2 * nW, wv216, nW, 1.f);
  cast_f32_to_bf16<<<cgW, cb, 0, stream>>>(wo,           wo16,  nW, 1.f);
  scale_f32<<<dim3(4), cb, 0, stream>>>(bin, bq_s, (long)D, qscale);

  const size_t smemG = 2 * 2 * BM * LDSS * sizeof(__bf16);            // 40 KB
  const size_t smemA = smemG + 3 * (size_t)BM * HDP * sizeof(__bf16); // ~142 KB

  // ---- stage 1: qk_in, v_in (fp32 x -> bf16 out) ----
  dim3 gg((unsigned)(M / BM), (unsigned)(D / BN));
  gemm_xwt<true, true><<<gg, cb, smemG, stream>>>((const void*)x, wqk16, bqk,
                                                  (void*)qkin, D, D);
  gemm_xwt<true, true><<<gg, cb, smemG, stream>>>((const void*)x, wv16, bvp,
                                                  (void*)vin16, D, D);

  // ---- stage 2: fused per-head q/k/v + masked segment softmax attention ----
  dim3 ga((unsigned)(M / BM), (unsigned)H);
  qkv_attn<<<ga, cb, smemA, stream>>>(qkin, vin16, wq16, wk16, wv216,
                                      bq_s, bin + D, bin + 2 * D, o16, D);

  // ---- stage 3: out-projection, fp32 output with bias ----
  gemm_xwt<false, false><<<gg, cb, smemG, stream>>>((const void*)o16, wo16, bo,
                                                    d_out, D, D);
}
